// VectorQuantizer_27943057228341
// MI455X (gfx1250) — matrix operand order repaired and checked by reading, not compiled, not run on hardware
//
#include <hip/hip_runtime.h>
#include <hip/hip_bf16.h>

typedef __attribute__((ext_vector_type(16))) _Float16 v16h;
typedef __attribute__((ext_vector_type(8)))  float    v8f;
typedef __attribute__((ext_vector_type(4)))  float    v4f;
typedef __attribute__((ext_vector_type(4)))  unsigned u4;

#define N_ROWS   65536      // 16 * 64 * 64 vectors
#define N_CODE   1024
#define EDIM     64
#define SPATIAL  4096       // 64*64 per image
#define CH_STRIDE 4096      // stride between channels in (B,C,H,W)
#define IMG_STRIDE 262144   // 64*4096 floats per image

// d_out layout (floats): [loss][z_q 4194304][perp][enc 67108864][idx 65536]
#define OFF_ZQ   1
#define OFF_PERP 4194305ull
#define OFF_ENC  4194306ull
#define OFF_IDX  71303170ull
#define ENC_ELEMS 67108864ll

// ---------------------------------------------------------------- prep ----
__global__ __launch_bounds__(256) void vq_prep(const float* __restrict__ emb,
                                               _Float16* __restrict__ embh,
                                               float* __restrict__ enorm,
                                               unsigned* __restrict__ counts,
                                               float* __restrict__ sse) {
    int t = blockIdx.x * 256 + threadIdx.x;          // 1024 threads
    if (t == 0) *sse = 0.0f;
    counts[t] = 0u;
    float s = 0.0f;
    #pragma unroll 8
    for (int d = 0; d < EDIM; ++d) {
        float v = emb[t * EDIM + d];
        s += v * v;
        embh[t * EDIM + d] = (_Float16)v;
    }
    enorm[t] = s;
}

// ------------------------------------------------------------- argmin ----
// One wave per 16-row tile; codebook staged through LDS in two 512-code
// phases (64 KB). Distances via v_wmma_f32_16x16x32_f16, K=64 in two chunks.
// B fragments and e-norms are software-pipelined one iteration ahead so the
// ds/load waits retire against loads from the previous iteration.
__global__ __launch_bounds__(256) void vq_argmin(const float* __restrict__ z,
                                                 const _Float16* __restrict__ embh,
                                                 const float* __restrict__ enorm,
                                                 int* __restrict__ idx,
                                                 unsigned* __restrict__ counts) {
    __shared__ __align__(32) _Float16 sB[512 * EDIM];   // 64 KB

    const int lane = threadIdx.x & 31;
    const int wave = threadIdx.x >> 5;
    const int rt   = blockIdx.x * 8 + wave;            // 16-row tile index
    const int hf   = lane >> 4;                        // K-half select
    const int l15  = lane & 15;
    const int row  = rt * 16 + l15;                    // A row for this lane
    const int bimg = row >> 12;
    const int spos = row & (SPATIAL - 1);
    const float* zp = z + (size_t)bimg * IMG_STRIDE + spos;

    // A fragments (contiguous k order, as B): element e -> VGPR e/2, hi/lo e&1
    //   K = kc*32 + hf*16 + e
    v16h a0, a1;
    #pragma unroll
    for (int e = 0; e < 16; ++e) {
        int kOff = hf * 16 + e;
        a0[e] = (_Float16)zp[(size_t)kOff * CH_STRIDE];
        a1[e] = (_Float16)zp[(size_t)(32 + kOff) * CH_STRIDE];
    }

    float best[8];
    int   bidx[8];
    #pragma unroll
    for (int v = 0; v < 8; ++v) { best[v] = 3.0e38f; bidx[v] = 0; }

    for (int p = 0; p < 2; ++p) {
        __syncthreads();
        { // cooperative 64 KB stage: 4096 x 16B
            const u4* src = (const u4*)(embh + (size_t)p * 512 * EDIM);
            u4* dst = (u4*)sB;
            for (int u = threadIdx.x; u < 4096; u += 256) dst[u] = src[u];
        }
        __syncthreads();

        // --- pipelined prologue: fragments/en for lct = 0 ---
        // B uses A's contiguous k order: lanes 0-15 hold K 0..15, lanes 16-31 K 16..31
        v16h b0 = *(const v16h*)&sB[l15 * EDIM + hf * 16];
        v16h b1 = *(const v16h*)&sB[l15 * EDIM + 32 + hf * 16];
        float en = enorm[p * 512 + l15];

        for (int lct = 0; lct < 32; ++lct) {
            // prefetch next iteration (wraps to 0 on last iter; values unused)
            const int ncl = ((lct + 1) & 31) * 16 + l15;
            v16h nb0 = *(const v16h*)&sB[ncl * EDIM + hf * 16];
            v16h nb1 = *(const v16h*)&sB[ncl * EDIM + 32 + hf * 16];
            float nen = enorm[p * 512 + ncl];

            v8f acc = {};
            acc = __builtin_amdgcn_wmma_f32_16x16x32_f16(false, a0, false, b0,
                                                         (short)0, acc, false, false);
            acc = __builtin_amdgcn_wmma_f32_16x16x32_f16(false, a1, false, b1,
                                                         (short)0, acc, false, false);
            const int col = p * 512 + lct * 16 + l15;
            #pragma unroll
            for (int v = 0; v < 8; ++v) {
                float sc = en - 2.0f * acc[v];         // ||e||^2 - 2 z.e
                if (sc < best[v]) { best[v] = sc; bidx[v] = col; }
            }
            b0 = nb0; b1 = nb1; en = nen;
        }
    }

    // per-row min across the 16 lanes that share M (C/D: lane = N, vgpr = M)
    #pragma unroll
    for (int v = 0; v < 8; ++v) {
        float bs = best[v]; int bi = bidx[v];
        #pragma unroll
        for (int off = 1; off <= 8; off <<= 1) {
            float os = __shfl_xor(bs, off, 32);
            int   oi = __shfl_xor(bi, off, 32);
            if (os < bs || (os == bs && oi < bi)) { bs = os; bi = oi; }
        }
        if (l15 == 0) {
            int r = rt * 16 + hf * 8 + v;
            idx[r] = bi;
            atomicAdd(&counts[bi], 1u);
        }
    }
}

// ------------------------------------------------------------- gather ----
__global__ __launch_bounds__(256) void vq_gather(const float* __restrict__ z,
                                                 const float* __restrict__ emb,
                                                 const int* __restrict__ idx,
                                                 float* __restrict__ zq,
                                                 float* __restrict__ sse) {
    const int n = blockIdx.x * 256 + threadIdx.x;      // 65536 rows
    const int bimg = n >> 12;
    const int spos = n & (SPATIAL - 1);
    const float* e  = emb + (size_t)idx[n] * EDIM;     // L2-resident gather
    const float* zp = z  + (size_t)bimg * IMG_STRIDE + spos;
    float*       op = zq + (size_t)bimg * IMG_STRIDE + spos;
    float acc = 0.0f;
    #pragma unroll 8
    for (int c = 0; c < EDIM; ++c) {
        float ev = e[c];
        float d  = ev - zp[(size_t)c * CH_STRIDE];
        acc += d * d;
        __builtin_nontemporal_store(ev, &op[(size_t)c * CH_STRIDE]);
    }
    #pragma unroll
    for (int off = 1; off < 32; off <<= 1) acc += __shfl_xor(acc, off, 32);
    if ((threadIdx.x & 31) == 0) atomicAdd(sse, acc);
}

// --------------------------------------------------------------- zero ----
// enc starts at float offset 4194306 (8-byte aligned only): write a 2-float
// head and tail, stream the 268MB middle with non-temporal B128 stores.
__global__ __launch_bounds__(256) void vq_zero(float* __restrict__ enc) {
    const long long n4 = (ENC_ELEMS - 4) / 4;          // 16777215 float4's
    v4f* __restrict__ p4 = (v4f*)(enc + 2);
    long long i = (long long)blockIdx.x * blockDim.x + threadIdx.x;
    const long long stride = (long long)gridDim.x * blockDim.x;
    v4f zv = {0.0f, 0.0f, 0.0f, 0.0f};
    for (; i < n4; i += stride) __builtin_nontemporal_store(zv, &p4[i]);
    if (blockIdx.x == 0 && threadIdx.x == 0) {
        enc[0] = 0.0f; enc[1] = 0.0f;
        enc[ENC_ELEMS - 2] = 0.0f; enc[ENC_ELEMS - 1] = 0.0f;
    }
}

// ------------------------------------------------------------ scatter ----
__global__ __launch_bounds__(256) void vq_scatter(const int* __restrict__ idx,
                                                  float* __restrict__ enc,
                                                  float* __restrict__ idxf) {
    const int n = blockIdx.x * 256 + threadIdx.x;      // 65536
    const int i = idx[n];
    enc[(size_t)n * N_CODE + i] = 1.0f;
    idxf[n] = (float)i;
}

// -------------------------------------------------------------- final ----
__global__ __launch_bounds__(1024) void vq_final(const unsigned* __restrict__ counts,
                                                 const float* __restrict__ sse,
                                                 float* __restrict__ out_loss,
                                                 float* __restrict__ out_perp) {
    __shared__ float red[1024];
    const int t = threadIdx.x;
    float em = (float)counts[t] * (1.0f / (float)N_ROWS);
    red[t] = em * logf(em + 1e-10f);
    __syncthreads();
    for (int s = 512; s > 0; s >>= 1) {
        if (t < s) red[t] += red[t + s];
        __syncthreads();
    }
    if (t == 0) {
        *out_perp = expf(-red[0]);
        *out_loss = 1.25f * (*sse) * (1.0f / 4194304.0f);  // (1+beta)*MSE
    }
}

// -------------------------------------------------------------- launch ---
extern "C" void kernel_launch(void* const* d_in, const int* in_sizes, int n_in,
                              void* d_out, int out_size, void* d_ws, size_t ws_size,
                              hipStream_t stream) {
    const float* z   = (const float*)d_in[0];   // (16,64,64,64) f32
    const float* emb = (const float*)d_in[1];   // (1024,64) f32
    float* out = (float*)d_out;

    char* ws = (char*)d_ws;                     // ~392 KB used
    int*      idx    = (int*)ws;                           // 262144 B
    unsigned* counts = (unsigned*)(ws + 262144);           //   4096 B
    float*    sse    = (float*)(ws + 266240);              //      4 B
    _Float16* embh   = (_Float16*)(ws + 266304);           // 131072 B
    float*    enorm  = (float*)(ws + 397376);              //   4096 B

    float* out_loss = out;
    float* zq       = out + OFF_ZQ;
    float* out_perp = out + OFF_PERP;
    float* enc      = out + OFF_ENC;
    float* idxf     = out + OFF_IDX;

    vq_prep<<<4, 256, 0, stream>>>(emb, embh, enorm, counts, sse);
    vq_argmin<<<512, 256, 0, stream>>>(z, embh, enorm, idx, counts);
    vq_gather<<<256, 256, 0, stream>>>(z, emb, idx, zq, sse);
    vq_zero<<<2048, 256, 0, stream>>>(enc);
    vq_scatter<<<256, 256, 0, stream>>>(idx, enc, idxf);
    vq_final<<<1, 1024, 0, stream>>>(counts, sse, out_loss, out_perp);
}
